// VGG_24464133718815
// MI455X (gfx1250) — matrix operand order repaired and checked by reading, not compiled, not run on hardware
//
#include <hip/hip_runtime.h>
#include <stdint.h>

typedef __attribute__((ext_vector_type(16))) _Float16 v16h;
typedef __attribute__((ext_vector_type(8)))  _Float16 v8h;
typedef __attribute__((ext_vector_type(8)))  float    v8f;

#define MT 128
#define NT 64
#define KT 32
#define LDSPAD 8   // row stride = 40 halves = 80B (16B-aligned rows & chunks)

// ---------------------------------------------------------------------------
// conv1: 3->64, 32x32, direct (K=27 too small for WMMA). fp32 NCHW in,
// f16 NHWC out [256,32,32,64].
// ---------------------------------------------------------------------------
__global__ __launch_bounds__(256)
void vgg_conv1_direct(const float* __restrict__ x, const float* __restrict__ w,
                      const float* __restrict__ b, _Float16* __restrict__ out) {
    __shared__ float lw[64 * 27];
    __shared__ float lb[64];
    const int tid = threadIdx.x;
    for (int i = tid; i < 64 * 27; i += 256) lw[i] = w[i];
    if (tid < 64) lb[tid] = b[tid];
    __syncthreads();

    const int g  = blockIdx.x * 256 + tid;     // 0..262143 = nb*1024 + y*32 + x
    const int xx = g & 31;
    const int yy = (g >> 5) & 31;
    const int nb = g >> 10;

    float vin[27];
#pragma unroll
    for (int cin = 0; cin < 3; ++cin)
#pragma unroll
        for (int dy = 0; dy < 3; ++dy)
#pragma unroll
            for (int dx = 0; dx < 3; ++dx) {
                const int iy = yy + dy - 1, ix = xx + dx - 1;
                float v = 0.0f;
                if (iy >= 0 && iy < 32 && ix >= 0 && ix < 32)
                    v = x[((nb * 3 + cin) * 32 + iy) * 32 + ix];
                vin[cin * 9 + dy * 3 + dx] = v;
            }

    _Float16* op = out + (size_t)g * 64;
    for (int co = 0; co < 64; ++co) {
        float acc = lb[co];
#pragma unroll
        for (int j = 0; j < 27; ++j) acc += vin[j] * lw[co * 27 + j];
        op[co] = (_Float16)acc;
    }
}

// ---------------------------------------------------------------------------
// 2x2 max-pool on f16 NHWC. Ho = output spatial, input is 2Ho x 2Ho.
// ---------------------------------------------------------------------------
__global__ __launch_bounds__(256)
void vgg_pool2(const _Float16* __restrict__ in, _Float16* __restrict__ out,
               int Ho, int C, int total) {
    const int o = blockIdx.x * 256 + threadIdx.x;
    if (o >= total) return;
    const int c = o % C;
    int s = o / C;
    const int xo = s % Ho; s /= Ho;
    const int yo = s % Ho;
    const int nb = s / Ho;
    const int H = Ho * 2, W = Ho * 2;
    const size_t base = (((size_t)(nb * H + yo * 2) * W) + xo * 2) * C + c;
    const float a = (float)in[base];
    const float b = (float)in[base + C];
    const float cc = (float)in[base + (size_t)W * C];
    const float d = (float)in[base + (size_t)W * C + C];
    out[o] = (_Float16)fmaxf(fmaxf(a, b), fmaxf(cc, d));
}

// ---------------------------------------------------------------------------
// Weight prep: fp32 OIHW [Cout,Cin,3,3] -> f16 [Cout][K], K=(dy*3+dx)*Cin+cin
// ---------------------------------------------------------------------------
__global__ __launch_bounds__(256)
void vgg_wprep(const float* __restrict__ w, _Float16* __restrict__ wt,
               int Cin, int total) {
    const int i = blockIdx.x * 256 + threadIdx.x;
    if (i >= total) return;
    const int K9 = Cin * 9;
    const int co = i / K9;
    const int rem = i % K9;             // = cin*9 + dydx in OIHW flat order
    const int cin = rem / 9;
    const int dydx = rem % 9;
    wt[(size_t)co * K9 + dydx * Cin + cin] = (_Float16)w[i];
}

// ---------------------------------------------------------------------------
// im2col GEMM conv via v_wmma_f32_16x16x32_f16, async-to-LDS staging.
//   inp : f16 NHWC [256,H,H,Cin]          (Cin multiple of 64)
//   wt  : f16 [Cout][K], K = 9*Cin, K-order (dy*3+dx)*Cin + cin
//   out : f32 NHWC [256,H,H,Cout] = conv + bias
// Block = 256 threads (8 waves) -> 128x64 output tile.
// Wave w owns M-groups {2*(w&3), 2*(w&3)+1} and N-groups {2*(w>>2), +1}
// -> 4 WMMAs per K step per wave.
// ---------------------------------------------------------------------------
__global__ __launch_bounds__(256)
void vgg_conv_gemm_wmma(const _Float16* __restrict__ inp,
                        const _Float16* __restrict__ wt,
                        const float* __restrict__ bias,
                        float* __restrict__ outp,
                        int H, int Cin, int Cout) {
    const int W    = H;
    const int Ktot = 9 * Cin;
    const int tid  = threadIdx.x;
    const int lane = tid & 31;
    const int wv   = tid >> 5;
    const int mBase = blockIdx.x * MT;
    const int nBase = blockIdx.y * NT;

    __shared__ __align__(16) _Float16 lA[MT][KT + LDSPAD];
    __shared__ __align__(16) _Float16 lB[NT][KT + LDSPAD];

    const int miB = (wv & 3) * 2;     // first of 2 M-groups for this wave
    const int ni0 = (wv >> 2) * 2;    // first of 2 N-groups for this wave

    v8f acc00 = {}, acc01 = {}, acc10 = {}, acc11 = {};
    const v8h vz = {};

    // --- staging assignments: 4 threads/row, 8 halves (16B) each ---
    const int rA0  = tid >> 2;            // A rows 0..63
    const int rA1  = rA0 + 64;            // A rows 64..127
    const int chkA = (tid & 3) * 8;
    const int rB   = tid >> 2;            // B rows 0..63
    const int chkB = (tid & 3) * 8;
    const int co   = nBase + rB;

    // decode output pixels for both staged A rows (loop-invariant)
    const int m0 = mBase + rA0;
    const int x0 = m0 % W, t0 = m0 / W, y0 = t0 % H, nb0 = t0 / H;
    const int m1 = mBase + rA1;
    const int x1 = m1 % W, t1 = m1 / W, y1 = t1 % H, nb1 = t1 / H;

    // LDS byte offsets for async destinations (generic-ptr low 32 bits)
    const unsigned ldsA0 = (unsigned)(uintptr_t)&lA[rA0][chkA];
    const unsigned ldsA1 = (unsigned)(uintptr_t)&lA[rA1][chkA];
    const unsigned ldsB  = (unsigned)(uintptr_t)&lB[rB][chkB];

    for (int dy = 0; dy < 3; ++dy) {
        const int iy0 = y0 + dy - 1;
        const int iy1 = y1 + dy - 1;
        for (int dx = 0; dx < 3; ++dx) {
            const int ix0 = x0 + dx - 1;
            const int ix1 = x1 + dx - 1;
            const bool inb0 = (iy0 >= 0) & (iy0 < H) & (ix0 >= 0) & (ix0 < W);
            const bool inb1 = (iy1 >= 0) & (iy1 < H) & (ix1 >= 0) & (ix1 < W);
            const _Float16* ga0 =
                inp + ((size_t)((nb0 * H + iy0) * W + ix0) * Cin + chkA);
            const _Float16* ga1 =
                inp + ((size_t)((nb1 * H + iy1) * W + ix1) * Cin + chkA);
            const _Float16* gb =
                wt + ((size_t)co * Ktot + (dy * 3 + dx) * Cin + chkB);

            for (int cin0 = 0; cin0 < Cin; cin0 += KT) {
                // --- async global->LDS staging (ASYNCcnt path) ---
                if (inb0)
                    asm volatile("global_load_async_to_lds_b128 %0, %1, off"
                                 :: "v"(ldsA0), "v"(ga0) : "memory");
                else
                    *(v8h*)&lA[rA0][chkA] = vz;
                if (inb1)
                    asm volatile("global_load_async_to_lds_b128 %0, %1, off"
                                 :: "v"(ldsA1), "v"(ga1) : "memory");
                else
                    *(v8h*)&lA[rA1][chkA] = vz;
                asm volatile("global_load_async_to_lds_b128 %0, %1, off"
                             :: "v"(ldsB), "v"(gb) : "memory");
                ga0 += KT; ga1 += KT; gb += KT;
                __builtin_prefetch(gb, 0, 1);   // next weight chunk

                asm volatile("s_wait_asynccnt 0" ::: "memory");
                __syncthreads();

                // --- fragments per ISA 7.12.2 layouts (b128 LDS reads) ---
                const int mla = lane & 15;
                const int kbA = (lane < 16) ? 0 : 8;
                const int kbB = (lane < 16) ? 0 : 8;

                const int mr0 = (miB + 0) * 16 + mla;
                v16h a0 = __builtin_shufflevector(
                    *(const v8h*)&lA[mr0][kbA], *(const v8h*)&lA[mr0][kbA + 16],
                    0, 1, 2, 3, 4, 5, 6, 7, 8, 9, 10, 11, 12, 13, 14, 15);
                const int mr1 = (miB + 1) * 16 + mla;
                v16h a1 = __builtin_shufflevector(
                    *(const v8h*)&lA[mr1][kbA], *(const v8h*)&lA[mr1][kbA + 16],
                    0, 1, 2, 3, 4, 5, 6, 7, 8, 9, 10, 11, 12, 13, 14, 15);

                const int nr0 = ni0 * 16 + mla;
                v16h b0 = __builtin_shufflevector(
                    *(const v8h*)&lB[nr0][kbB], *(const v8h*)&lB[nr0][kbB + 16],
                    0, 1, 2, 3, 4, 5, 6, 7, 8, 9, 10, 11, 12, 13, 14, 15);
                const int nr1 = nr0 + 16;
                v16h b1 = __builtin_shufflevector(
                    *(const v8h*)&lB[nr1][kbB], *(const v8h*)&lB[nr1][kbB + 16],
                    0, 1, 2, 3, 4, 5, 6, 7, 8, 9, 10, 11, 12, 13, 14, 15);

                acc00 = __builtin_amdgcn_wmma_f32_16x16x32_f16(
                    false, a0, false, b0, (short)0, acc00, false, false);
                acc01 = __builtin_amdgcn_wmma_f32_16x16x32_f16(
                    false, a0, false, b1, (short)0, acc01, false, false);
                acc10 = __builtin_amdgcn_wmma_f32_16x16x32_f16(
                    false, a1, false, b0, (short)0, acc10, false, false);
                acc11 = __builtin_amdgcn_wmma_f32_16x16x32_f16(
                    false, a1, false, b1, (short)0, acc11, false, false);

                __syncthreads();
            }
        }
    }

    // --- epilogue: C/D layout: vgpr r -> M = r + 8*(lane>=16), N = lane&15 ---
    const int hi8 = (lane < 16) ? 0 : 8;
    const int mo0 = mBase + (miB + 0) * 16 + hi8;
    const int mo1 = mBase + (miB + 1) * 16 + hi8;
    const int nc0 = nBase + ni0 * 16 + (lane & 15);
    const int nc1 = nc0 + 16;
    const float bv0 = bias[nc0];
    const float bv1 = bias[nc1];
#pragma unroll
    for (int r = 0; r < 8; ++r) {
        outp[(size_t)(mo0 + r) * Cout + nc0] = acc00[r] + bv0;
        outp[(size_t)(mo0 + r) * Cout + nc1] = acc01[r] + bv1;
        outp[(size_t)(mo1 + r) * Cout + nc0] = acc10[r] + bv0;
        outp[(size_t)(mo1 + r) * Cout + nc1] = acc11[r] + bv1;
    }
}

// ---------------------------------------------------------------------------
// BN batch statistics -> per-channel scale/shift. One block per channel.
// ---------------------------------------------------------------------------
__global__ __launch_bounds__(256)
void vgg_bn_stats(const float* __restrict__ x, const float* __restrict__ g,
                  const float* __restrict__ be, float* __restrict__ scale,
                  float* __restrict__ shift, int M, int C) {
    const int c = blockIdx.x;
    __shared__ float s1[256];
    __shared__ float s2[256];
    float a = 0.0f, b2 = 0.0f;
    for (int m = threadIdx.x; m < M; m += 256) {
        const float v = x[(size_t)m * C + c];
        a += v;
        b2 += v * v;
    }
    s1[threadIdx.x] = a;
    s2[threadIdx.x] = b2;
    __syncthreads();
    for (int st = 128; st > 0; st >>= 1) {
        if (threadIdx.x < st) {
            s1[threadIdx.x] += s1[threadIdx.x + st];
            s2[threadIdx.x] += s2[threadIdx.x + st];
        }
        __syncthreads();
    }
    if (threadIdx.x == 0) {
        const float mu  = s1[0] / (float)M;
        const float var = s2[0] / (float)M - mu * mu;
        const float sc  = g[c] * rsqrtf(var + 1e-5f);
        scale[c] = sc;
        shift[c] = be[c] - mu * sc;
    }
}

// ---------------------------------------------------------------------------
// BN apply + ReLU, f32 -> f16 (next layer input)
// ---------------------------------------------------------------------------
__global__ __launch_bounds__(256)
void vgg_bn_relu(const float* __restrict__ x, const float* __restrict__ scale,
                 const float* __restrict__ shift, _Float16* __restrict__ y,
                 int C, int total) {
    const int i = blockIdx.x * 256 + threadIdx.x;
    if (i >= total) return;
    const int c = i % C;
    float v = x[i] * scale[c] + shift[c];
    v = v > 0.0f ? v : 0.0f;
    y[i] = (_Float16)v;
}

// ---------------------------------------------------------------------------
// Final FC: [256,512] f16 @ fc_w^T [512,10] fp32 + fc_b -> f32 [256,10]
// ---------------------------------------------------------------------------
__global__ __launch_bounds__(256)
void vgg_fc10(const _Float16* __restrict__ act, const float* __restrict__ w,
              const float* __restrict__ b, float* __restrict__ out) {
    const int i = blockIdx.x * 256 + threadIdx.x;
    if (i >= 2560) return;
    const int nb = i / 10, cls = i % 10;
    const _Float16* a = act + (size_t)nb * 512;
    const float* wr = w + (size_t)cls * 512;
    float acc = b[cls];
    for (int k = 0; k < 512; ++k) acc += (float)a[k] * wr[k];
    out[i] = acc;
}

// ---------------------------------------------------------------------------
extern "C" void kernel_launch(void* const* d_in, const int* in_sizes, int n_in,
                              void* d_out, int out_size, void* d_ws, size_t ws_size,
                              hipStream_t stream) {
    (void)in_sizes; (void)n_in; (void)out_size; (void)ws_size;

    const float* x   = (const float*)d_in[0];
    const float* c1w = (const float*)d_in[1];
    const float* c1b = (const float*)d_in[2];
    const float* fcw = (const float*)d_in[31];
    const float* fcb = (const float*)d_in[32];

    // workspace carve-up (all 256B-aligned offsets)
    char* ws = (char*)d_ws;
    _Float16* act0  = (_Float16*)(ws);                 // 33,554,432 B
    _Float16* act1  = (_Float16*)(ws + 33554432);      // 33,554,432 B
    float*    convb = (float*)   (ws + 67108864);      // 33,554,432 B
    _Float16* wt    = (_Float16*)(ws + 100663296);     //  4,718,592 B
    float*    scale = (float*)   (ws + 105381888);     //      2,048 B
    float*    shift = scale + 512;

    // conv1 -> f16 NHWC [256,32,32,64]
    vgg_conv1_direct<<<1024, 256, 0, stream>>>(x, c1w, c1b, act0);

    struct Layer { int Hin; int pool; int Cin; int Cout; };
    const Layer L[7] = {
        {32, 1,  64, 128}, {16, 1, 128, 256}, { 8, 0, 256, 256},
        { 8, 1, 256, 512}, { 4, 0, 512, 512}, { 4, 1, 512, 512},
        { 2, 0, 512, 512}};

    _Float16* curAct   = act0;
    _Float16* otherAct = act1;

    for (int i = 0; i < 7; ++i) {
        const float* w  = (const float*)d_in[3 + 4 * i + 0];
        const float* bb = (const float*)d_in[3 + 4 * i + 1];
        const float* g  = (const float*)d_in[3 + 4 * i + 2];
        const float* be = (const float*)d_in[3 + 4 * i + 3];

        const int Cin  = L[i].Cin;
        const int Cout = L[i].Cout;
        const int H    = L[i].pool ? L[i].Hin / 2 : L[i].Hin;

        _Float16* convIn;
        _Float16* dst;
        if (L[i].pool) {
            const int ptot = 256 * H * H * Cin;
            vgg_pool2<<<(ptot + 255) / 256, 256, 0, stream>>>(curAct, otherAct,
                                                              H, Cin, ptot);
            convIn = otherAct;
            dst = curAct;       // next layer input stays in curAct
        } else {
            convIn = curAct;
            dst = otherAct;
        }

        const int wtot = Cout * Cin * 9;
        vgg_wprep<<<(wtot + 255) / 256, 256, 0, stream>>>(w, wt, Cin, wtot);

        const int M = 256 * H * H;           // always a multiple of 128
        dim3 grid(M / MT, Cout / NT);
        vgg_conv_gemm_wmma<<<grid, 256, 0, stream>>>(convIn, wt, bb, convb,
                                                     H, Cin, Cout);

        vgg_bn_stats<<<Cout, 256, 0, stream>>>(convb, g, be, scale, shift,
                                               M, Cout);
        const int tot = M * Cout;
        vgg_bn_relu<<<(tot + 255) / 256, 256, 0, stream>>>(convb, scale, shift,
                                                           dst, Cout, tot);
        if (!L[i].pool) {
            _Float16* t = curAct; curAct = otherAct; otherAct = t;
        }
    }

    // final 2x2 pool -> [256,512], then FC -> d_out [256,10] f32
    const int ftot = 256 * 512;
    vgg_pool2<<<(ftot + 255) / 256, 256, 0, stream>>>(curAct, otherAct,
                                                      1, 512, ftot);
    vgg_fc10<<<10, 256, 0, stream>>>(otherAct, fcw, fcb, (float*)d_out);
}
